// model_class_49082886259430
// MI455X (gfx1250) — matrix operand order repaired and checked by reading, not compiled, not run on hardware
//
#include <hip/hip_runtime.h>

// ---------------------------------------------------------------------------
// Seq2seq (bidir LSTM encoder x2 + attention decoder), batch 1.
//   I=27, H=1024, 4H=4096, T=256, O=128, steps=30
// MI455X strategy:
//   * All large weight matrices cast once to bf16 (RNE) in workspace: halves
//     the per-step weight streaming that dominates both the sequential
//     recurrent critical path (L2-resident whh) and the decoder (HBM-streamed
//     d0/d1, ~270MB/step f32 -> ~135MB/step bf16). f32 accumulation throughout.
//   * GEMM-shaped work (layer-1 input projection, attention u) ->
//     V_WMMA_F32_16X16X32_BF16 (wave32 WMMA, K=32 per instruction).
//   * Recurrences / decoder -> persistent kernels, device-wide sense barrier,
//     wave-per-row coalesced packed-bf16 matvecs, per-block (h,c) in LDS.
// ---------------------------------------------------------------------------

typedef __attribute__((ext_vector_type(16))) __bf16 v16bf;
typedef __attribute__((ext_vector_type(8)))  float  v8f;

static constexpr int NTH   = 256;           // threads per block (8 wave32)
static constexpr int NBLK  = 64;            // persistent grid size
static constexpr int NWAVE = NBLK * NTH / 32;
static constexpr int HD = 1024;             // hidden
static constexpr int G4 = 4096;             // 4*H (gates)
static constexpr int TS = 256;              // sequence length
static constexpr int OD = 128;              // output vocab

__device__ __forceinline__ float sigmoidf_(float x) { return 1.f / (1.f + expf(-x)); }

// unpack two packed bf16 (in one dword) to two floats
__device__ __forceinline__ float2 bf2f2(unsigned u) {
  float2 r;
  r.x = __uint_as_float(u << 16);
  r.y = __uint_as_float(u & 0xffff0000u);
  return r;
}

__device__ __forceinline__ float wave_reduce(float a) {
#pragma unroll
  for (int off = 16; off; off >>= 1) a += __shfl_down(a, off, 32);
  return a;  // valid in lane 0
}

// Device-wide sense barrier (persistent kernels; grid <= resident capacity).
__device__ __forceinline__ void grid_sync(unsigned* cnt, unsigned* gen) {
  __syncthreads();
  if (threadIdx.x == 0) {
    __threadfence();
    unsigned g = __atomic_load_n(gen, __ATOMIC_ACQUIRE);
    if (atomicAdd(cnt, 1u) == gridDim.x - 1u) {
      atomicExch(cnt, 0u);
      __threadfence();
      atomicAdd(gen, 1u);
    } else {
      while (__atomic_load_n(gen, __ATOMIC_ACQUIRE) == g)
        __builtin_amdgcn_s_sleep(1);
    }
    __threadfence();
  }
  __syncthreads();
}

__global__ void k_init(unsigned* bar) {
  if (threadIdx.x < 2) bar[threadIdx.x] = 0u;
}

// f32 -> bf16 (round-to-nearest-even), 2 elements per thread-iteration.
__global__ void k_tobf16(const float* __restrict__ s, unsigned* __restrict__ d,
                         long npairs) {
  long i = (long)blockIdx.x * blockDim.x + threadIdx.x;
  const long stride = (long)gridDim.x * blockDim.x;
  for (; i < npairs; i += stride) {
    float2 f = ((const float2*)s)[i];
    unsigned lo = __float_as_uint(f.x);
    unsigned hi = __float_as_uint(f.y);
    lo += 0x7fffu + ((lo >> 16) & 1u);
    hi += 0x7fffu + ((hi >> 16) & 1u);
    d[i] = (lo >> 16) | (hi & 0xffff0000u);
  }
}

// g0[d][t][r] = e0_bih[d][r] + e0_bhh[d][r] + sum_k e0_wih[d][r][k] * x[t][k]   (K=27)
__global__ void k_g0(const float* __restrict__ x, const float* __restrict__ wih,
                     const float* __restrict__ bih, const float* __restrict__ bhh,
                     float* __restrict__ g0) {
  __shared__ float xl[32];
  const int d = blockIdx.x >> 8, t = blockIdx.x & 255;
  if (threadIdx.x < 27) xl[threadIdx.x] = x[t * 27 + threadIdx.x];
  __syncthreads();
  for (int r = threadIdx.x; r < G4; r += NTH) {
    const float* wr = wih + ((size_t)d * G4 + r) * 27;
    float acc = bih[d * G4 + r] + bhh[d * G4 + r];
#pragma unroll
    for (int k = 0; k < 27; ++k) acc += wr[k] * xl[k];
    g0[((size_t)d * TS + t) * G4 + r] = acc;
  }
}

// C[m][n] = b1[n] (+ b2[n]) + sum_k A[m][k] * W[n][k], A/W bf16, C f32.
// V_WMMA_F32_16X16X32_BF16; wave owns a 16x16 C tile.
// ISA fragment layouts (CDNA5 ISA, 16-bit operands):
//   A (16x32): lane holds row m=lane&15; elements [0..7] = K = k0+8*half+0..7,
//              [8..15] = K = k0+16+8*half+0..7      (half = lane>=16)
//   B (32x16): lane holds col n=lane&15; same K mapping as A
//   C/D: vgpr v -> row v + 8*half, col lane&15
__global__ void k_gemm_bf16(const unsigned short* __restrict__ A,
                            const unsigned short* __restrict__ W,
                            const float* __restrict__ b1, const float* __restrict__ b2,
                            float* __restrict__ C, int M, int N, int K,
                            long Wz, long Bz, long Cz) {
  const unsigned short* Wp = W + (size_t)blockIdx.z * Wz;
  const float* b1p = b1 + (size_t)blockIdx.z * Bz;
  const float* b2p = b2 ? b2 + (size_t)blockIdx.z * Bz : nullptr;
  float* Cp = C + (size_t)blockIdx.z * Cz;
  const int lane = threadIdx.x & 31, wave = threadIdx.x >> 5;
  const int mt = blockIdx.y * 16;
  const int nt = (blockIdx.x * 8 + wave) * 16;
  if (nt >= N || mt >= M) return;
  const int half = lane >> 4;
  const int mr = mt + (lane & 15);
  const int nr = nt + (lane & 15);
  const unsigned short* Ar = A + (size_t)mr * K;
  const unsigned short* Wr = Wp + (size_t)nr * K;
  v8f acc = {};
  for (int k0 = 0; k0 < K; k0 += 32) {
    union { v16bf v; uint4 q[2]; } ua, ub;
    ua.q[0] = *(const uint4*)(Ar + k0 + 8 * half);
    ua.q[1] = *(const uint4*)(Ar + k0 + 16 + 8 * half);
    ub.q[0] = *(const uint4*)(Wr + k0 + 8 * half);
    ub.q[1] = *(const uint4*)(Wr + k0 + 16 + 8 * half);
    acc = __builtin_amdgcn_wmma_f32_16x16x32_bf16(false, ua.v, false, ub.v,
                                                  (short)0, acc, false, false);
  }
  const float bb = b1p[nr] + (b2p ? b2p[nr] : 0.f);
  const int mbase = mt + 8 * half;
#pragma unroll
  for (int v = 0; v < 8; ++v)
    Cp[(size_t)(mbase + v) * N + nr] = acc[v] + bb;
}

// dot(bf16-row, f32-vector) over a wave: 4 packed weights per lane-load.
__device__ __forceinline__ float wave_dot_bf(const unsigned short* __restrict__ w,
                                             const float* __restrict__ x,
                                             int K, int lane) {
  const uint2* w4 = (const uint2*)w;
  float acc = 0.f;
  for (int q = lane; q < (K >> 2); q += 32) {
    const uint2 p = w4[q];
    const float2 a = bf2f2(p.x), b = bf2f2(p.y);
    const int k = q << 2;
    acc += a.x * x[k] + a.y * x[k + 1] + b.x * x[k + 2] + b.y * x[k + 3];
  }
  return acc;
}

// Persistent bidirectional LSTM layer. gpre holds wih@x + biases for both dirs.
// One grid barrier per timestep; gate scratch double-buffered on t parity; each
// block keeps its own (h,c) copy in LDS and applies the nonlinearity redundantly.
__global__ void __launch_bounds__(NTH)
k_lstm_layer(const float* __restrict__ gpre,            // [2][T][4096]
             const unsigned short* __restrict__ whh,    // [2][4096][1024] bf16
             float* __restrict__ out,                   // [T][2048] (fwd | bwd)
             float* __restrict__ hT,                    // [2][1024] final hidden
             float* __restrict__ gsc,                   // [2 parity][2][4096]
             unsigned* bar, int Tn) {
  __shared__ float hs[2][HD];
  __shared__ float cs[2][HD];
  for (int i = threadIdx.x; i < HD; i += NTH) {
    hs[0][i] = 0.f; hs[1][i] = 0.f; cs[0][i] = 0.f; cs[1][i] = 0.f;
  }
  __syncthreads();
  const int lane = threadIdx.x & 31;
  const int wgid = blockIdx.x * (NTH / 32) + (threadIdx.x >> 5);
  unsigned *cnt = bar, *gen = bar + 1;

  for (int t = 0; t < Tn; ++t) {
    float* gout = gsc + (size_t)(t & 1) * 2 * G4;
    // recurrent matvec: 8192 rows over 512 waves, bf16 weights
    for (int row = wgid; row < 2 * G4; row += NWAVE) {
      const int d = row >> 12, r = row & (G4 - 1);
      float acc = wave_dot_bf(whh + ((size_t)d * G4 + r) * HD, hs[d], HD, lane);
      acc = wave_reduce(acc);
      if (lane == 0) {
        const int tc = d ? (Tn - 1 - t) : t;
        gout[d * G4 + r] = acc + gpre[((size_t)d * Tn + tc) * G4 + r];
      }
    }
    grid_sync(cnt, gen);
    // nonlinearity: every block updates its private (h,c); block 0 emits output
    for (int j = threadIdx.x; j < HD; j += NTH) {
#pragma unroll
      for (int d = 0; d < 2; ++d) {
        const float ig = gout[d * G4 + j];
        const float fg = gout[d * G4 + HD + j];
        const float gg = gout[d * G4 + 2 * HD + j];
        const float og = gout[d * G4 + 3 * HD + j];
        const float cn = sigmoidf_(fg) * cs[d][j] + sigmoidf_(ig) * tanhf(gg);
        const float hn = sigmoidf_(og) * tanhf(cn);
        cs[d][j] = cn; hs[d][j] = hn;
        if (blockIdx.x == 0) {
          const int tc = d ? (Tn - 1 - t) : t;
          out[(size_t)tc * (2 * HD) + d * HD + j] = hn;
        }
      }
    }
    __syncthreads();
  }
  if (blockIdx.x == 0 && hT)
    for (int j = threadIdx.x; j < HD; j += NTH) {
      hT[j] = hs[0][j]; hT[HD + j] = hs[1][j];
    }
}

// Persistent attention decoder: 30 steps, ~10 grid barriers per step.
// d0/d1 weight matrices are bf16 (HBM-streamed every step -> halved traffic).
__global__ void __launch_bounds__(NTH)
k_decoder(const float* __restrict__ encout, const float* __restrict__ uatt,
          const float* __restrict__ h0T, const float* __restrict__ h1T,
          const float* __restrict__ w1_w, const float* __restrict__ w1_b,
          const float* __restrict__ w2_w, const float* __restrict__ w2_b,
          const float* __restrict__ w3_w, const float* __restrict__ w3_b,
          const float* __restrict__ w4_w, const float* __restrict__ w4_b,
          const float* __restrict__ w_w,  const float* __restrict__ w_b,
          const float* __restrict__ att_w, const float* __restrict__ att_b,
          const float* __restrict__ o2c_w, const float* __restrict__ o2c_b,
          const unsigned short* __restrict__ d0_wih,
          const unsigned short* __restrict__ d0_whh,
          const float* __restrict__ d0_bih, const float* __restrict__ d0_bhh,
          const unsigned short* __restrict__ d1_wih,
          const unsigned short* __restrict__ d1_whh,
          const float* __restrict__ d1_bih, const float* __restrict__ d1_bhh,
          const float* __restrict__ ho_w, const float* __restrict__ ho_b,
          float* dsh, float* dsc, float* wsum, float* wp, float* vatt,
          float* xt, float* out0g, float* outg, float* logits, float* decin,
          float* dgates, float* d_out, unsigned* bar, int steps) {
  __shared__ float sxt[G4];        // staged xt / out0 (16KB)
  __shared__ float sh2[2 * HD];    // staged hidden pair (8KB)
  __shared__ float sv[TS];         // attention weights
  __shared__ float red[NTH];
  __shared__ float slog[OD];
  __shared__ int sidx;
  const int tid  = threadIdx.x;
  const int lane = tid & 31;
  const int wgid = blockIdx.x * (NTH / 32) + (tid >> 5);
  const int gt   = blockIdx.x * NTH + tid;
  unsigned *cnt = bar, *gen = bar + 1;

  // init decoder state: h = [h0T_f, h0T_b, h1T_f, h1T_b], c = 0, dec_in = 0
  if (gt < HD) {
    dsh[0 * HD + gt] = h0T[gt];      dsh[1 * HD + gt] = h0T[HD + gt];
    dsh[2 * HD + gt] = h1T[gt];      dsh[3 * HD + gt] = h1T[HD + gt];
    dsc[0 * HD + gt] = 0.f; dsc[1 * HD + gt] = 0.f;
    dsc[2 * HD + gt] = 0.f; dsc[3 * HD + gt] = 0.f;
  }
  if (gt < OD) decin[gt] = 0.f;
  grid_sync(cnt, gen);

  for (int step = 0; step < steps; ++step) {
    // S1: wsum = sum_m (wm_w @ h[m] + wm_b)
    for (int r = wgid; r < HD; r += NWAVE) {
      const float* r1 = w1_w + (size_t)r * HD;
      const float* r2 = w2_w + (size_t)r * HD;
      const float* r3 = w3_w + (size_t)r * HD;
      const float* r4 = w4_w + (size_t)r * HD;
      float acc = 0.f;
      for (int k = lane; k < HD; k += 32)
        acc += r1[k] * dsh[k] + r2[k] * dsh[HD + k] +
               r3[k] * dsh[2 * HD + k] + r4[k] * dsh[3 * HD + k];
      acc = wave_reduce(acc);
      if (lane == 0) wsum[r] = acc + w1_b[r] + w2_b[r] + w3_b[r] + w4_b[r];
    }
    grid_sync(cnt, gen);
    // S2: wp = w_w @ wsum + w_b ; xt[0:2H] = o2c_w @ dec_in + o2c_b
    for (int r = wgid; r < 3 * HD; r += NWAVE) {
      if (r < HD) {
        const float* wr = w_w + (size_t)r * HD;
        float acc = 0.f;
        for (int k = lane; k < HD; k += 32) acc += wr[k] * wsum[k];
        acc = wave_reduce(acc);
        if (lane == 0) wp[r] = acc + w_b[r];
      } else {
        const int rr = r - HD;
        const float* wr = o2c_w + (size_t)rr * OD;
        float acc = 0.f;
        for (int k = lane; k < OD; k += 32) acc += wr[k] * decin[k];
        acc = wave_reduce(acc);
        if (lane == 0) xt[rr] = acc + o2c_b[rr];
      }
    }
    grid_sync(cnt, gen);
    // S3: v[t] = tanh(u[t] + wp) . att_w + att_b
    for (int r = wgid; r < TS; r += NWAVE) {
      const float* ur = uatt + (size_t)r * HD;
      float acc = 0.f;
      for (int k = lane; k < HD; k += 32) acc += tanhf(ur[k] + wp[k]) * att_w[k];
      acc = wave_reduce(acc);
      if (lane == 0) vatt[r] = acc + att_b[0];
    }
    grid_sync(cnt, gen);
    // S5: per-block redundant softmax(v) in LDS, then context cj -> xt[2H:4H]
    sv[tid] = vatt[tid];             // NTH == TS == 256
    red[tid] = sv[tid];
    __syncthreads();
    for (int s = NTH / 2; s > 0; s >>= 1) {
      if (tid < s) red[tid] = fmaxf(red[tid], red[tid + s]);
      __syncthreads();
    }
    const float vmax = red[0];
    __syncthreads();
    const float ev = expf(sv[tid] - vmax);
    red[tid] = ev;
    __syncthreads();
    for (int s = NTH / 2; s > 0; s >>= 1) {
      if (tid < s) red[tid] += red[tid + s];
      __syncthreads();
    }
    const float vsum = red[0];
    __syncthreads();
    sv[tid] = ev / vsum;
    __syncthreads();
    if (gt < 2 * HD) {
      float acc = 0.f;
      for (int t2 = 0; t2 < TS; ++t2) acc += sv[t2] * encout[(size_t)t2 * (2 * HD) + gt];
      xt[2 * HD + gt] = acc;
    }
    grid_sync(cnt, gen);
    // S6: layer-0 decoder gates (K = 4096 + 1024), bf16 weights, LDS-staged acts
    for (int i = tid; i < G4; i += NTH) sxt[i] = xt[i];
    for (int i = tid; i < 2 * HD; i += NTH) sh2[i] = dsh[i];
    __syncthreads();
    for (int row = wgid; row < 2 * G4; row += NWAVE) {
      const int cell = row >> 12, r = row & (G4 - 1);
      float acc = wave_dot_bf(d0_wih + ((size_t)cell * G4 + r) * G4, sxt, G4, lane);
      acc += wave_dot_bf(d0_whh + ((size_t)cell * G4 + r) * HD, sh2 + cell * HD, HD, lane);
      acc = wave_reduce(acc);
      if (lane == 0)
        dgates[cell * G4 + r] = acc + d0_bih[cell * G4 + r] + d0_bhh[cell * G4 + r];
    }
    grid_sync(cnt, gen);
    // S7: update cells 0,1 ; out0 = concat(h0f, h0b)
    if (gt < 2 * HD) {
      const int cell = gt >> 10, j = gt & (HD - 1);
      const float ig = dgates[cell * G4 + j];
      const float fg = dgates[cell * G4 + HD + j];
      const float gg = dgates[cell * G4 + 2 * HD + j];
      const float og = dgates[cell * G4 + 3 * HD + j];
      const float cn = sigmoidf_(fg) * dsc[cell * HD + j] + sigmoidf_(ig) * tanhf(gg);
      const float hn = sigmoidf_(og) * tanhf(cn);
      dsc[cell * HD + j] = cn; dsh[cell * HD + j] = hn; out0g[cell * HD + j] = hn;
    }
    grid_sync(cnt, gen);
    // S8: layer-1 decoder gates (K = 2048 + 1024), bf16 weights
    for (int i = tid; i < 2 * HD; i += NTH) sxt[i] = out0g[i];
    for (int i = tid; i < 2 * HD; i += NTH) sh2[i] = dsh[2 * HD + i];
    __syncthreads();
    for (int row = wgid; row < 2 * G4; row += NWAVE) {
      const int cell = row >> 12, r = row & (G4 - 1);
      float acc = wave_dot_bf(d1_wih + ((size_t)cell * G4 + r) * (2 * HD), sxt, 2 * HD, lane);
      acc += wave_dot_bf(d1_whh + ((size_t)cell * G4 + r) * HD, sh2 + cell * HD, HD, lane);
      acc = wave_reduce(acc);
      if (lane == 0)
        dgates[cell * G4 + r] = acc + d1_bih[cell * G4 + r] + d1_bhh[cell * G4 + r];
    }
    grid_sync(cnt, gen);
    // S9: update cells 2,3 ; out = concat(h1f, h1b)
    if (gt < 2 * HD) {
      const int cell = gt >> 10, j = gt & (HD - 1);
      const float ig = dgates[cell * G4 + j];
      const float fg = dgates[cell * G4 + HD + j];
      const float gg = dgates[cell * G4 + 2 * HD + j];
      const float og = dgates[cell * G4 + 3 * HD + j];
      const float cn = sigmoidf_(fg) * dsc[(2 + cell) * HD + j] + sigmoidf_(ig) * tanhf(gg);
      const float hn = sigmoidf_(og) * tanhf(cn);
      dsc[(2 + cell) * HD + j] = cn; dsh[(2 + cell) * HD + j] = hn;
      outg[cell * HD + j] = hn;
    }
    grid_sync(cnt, gen);
    // S10: logits = ho_w @ out + ho_b
    for (int r = wgid; r < OD; r += NWAVE) {
      const float* wr = ho_w + (size_t)r * (2 * HD);
      float acc = 0.f;
      for (int k = lane; k < 2 * HD; k += 32) acc += wr[k] * outg[k];
      acc = wave_reduce(acc);
      if (lane == 0) logits[r] = acc + ho_b[r];
    }
    grid_sync(cnt, gen);
    // S11: log_softmax + argmax -> one-hot feedback (block 0 writes globals)
    if (tid < OD) slog[tid] = logits[tid];
    if (tid == 0) sidx = OD;
    red[tid] = (tid < OD) ? logits[tid] : -3.4e38f;
    __syncthreads();
    for (int s = NTH / 2; s > 0; s >>= 1) {
      if (tid < s) red[tid] = fmaxf(red[tid], red[tid + s]);
      __syncthreads();
    }
    const float lmax = red[0];
    __syncthreads();
    red[tid] = (tid < OD) ? expf(slog[tid] - lmax) : 0.f;
    __syncthreads();
    for (int s = NTH / 2; s > 0; s >>= 1) {
      if (tid < s) red[tid] += red[tid + s];
      __syncthreads();
    }
    const float lse = lmax + logf(red[0]);
    __syncthreads();
    if (tid < OD && slog[tid] == lmax) atomicMin(&sidx, tid);
    __syncthreads();
    if (blockIdx.x == 0 && tid < OD) {
      d_out[(size_t)step * OD + tid] = slog[tid] - lse;
      decin[tid] = (tid == sidx) ? 1.f : 0.f;
    }
    grid_sync(cnt, gen);
  }
}

extern "C" void kernel_launch(void* const* d_in, const int* in_sizes, int n_in,
                              void* d_out, int out_size, void* d_ws, size_t ws_size,
                              hipStream_t stream) {
  const float* inputs = (const float*)d_in[0];
  // d_in[1] = max_chars (derive steps from out_size instead)
  const float* e0_wih = (const float*)d_in[2];
  const float* e0_whh = (const float*)d_in[3];
  const float* e0_bih = (const float*)d_in[4];
  const float* e0_bhh = (const float*)d_in[5];
  const float* e1_wih = (const float*)d_in[6];
  const float* e1_whh = (const float*)d_in[7];
  const float* e1_bih = (const float*)d_in[8];
  const float* e1_bhh = (const float*)d_in[9];
  const float* d0_wih = (const float*)d_in[10];
  const float* d0_whh = (const float*)d_in[11];
  const float* d0_bih = (const float*)d_in[12];
  const float* d0_bhh = (const float*)d_in[13];
  const float* d1_wih = (const float*)d_in[14];
  const float* d1_whh = (const float*)d_in[15];
  const float* d1_bih = (const float*)d_in[16];
  const float* d1_bhh = (const float*)d_in[17];
  const float* u_w  = (const float*)d_in[18];
  const float* u_b  = (const float*)d_in[19];
  const float* w1_w = (const float*)d_in[20];
  const float* w1_b = (const float*)d_in[21];
  const float* w2_w = (const float*)d_in[22];
  const float* w2_b = (const float*)d_in[23];
  const float* w3_w = (const float*)d_in[24];
  const float* w3_b = (const float*)d_in[25];
  const float* w4_w = (const float*)d_in[26];
  const float* w4_b = (const float*)d_in[27];
  const float* w_w  = (const float*)d_in[28];
  const float* w_b  = (const float*)d_in[29];
  const float* att_w = (const float*)d_in[30];
  const float* att_b = (const float*)d_in[31];
  const float* o2c_w = (const float*)d_in[32];
  const float* o2c_b = (const float*)d_in[33];
  const float* ho_w = (const float*)d_in[34];
  const float* ho_b = (const float*)d_in[35];
  float* out = (float*)d_out;
  const int steps = out_size / OD;  // = max_chars

  // Workspace carve-up: ~23 MB f32 scratch + ~208 MB bf16 weight cache.
  float* wsf = (float*)d_ws;
  size_t off = 0;
  auto alloc = [&](size_t nfloats) {
    float* p = wsf + off;
    off += (nfloats + 63) & ~(size_t)63;
    return p;
  };
  // f32 scratch
  float* g0     = alloc((size_t)2 * TS * G4);   // wih@x + biases, layer 0
  float* g1     = alloc((size_t)2 * TS * G4);   // wih@h0 + biases, layer 1 (WMMA)
  float* h0out  = alloc((size_t)TS * 2 * HD);
  float* encout = alloc((size_t)TS * 2 * HD);
  float* uatt   = alloc((size_t)TS * HD);       // enc @ u_w.T + u_b (WMMA)
  float* gsc    = alloc((size_t)2 * 2 * G4);    // double-buffered gate scratch
  float* h0T    = alloc(2 * HD);
  float* h1T    = alloc(2 * HD);
  float* wsum   = alloc(HD);
  float* wp     = alloc(HD);
  float* vatt   = alloc(TS);
  float* xt     = alloc(G4);
  float* out0g  = alloc(2 * HD);
  float* outg   = alloc(2 * HD);
  float* logits = alloc(OD);
  float* decin  = alloc(OD);
  float* dgates = alloc((size_t)2 * G4);
  float* dsh    = alloc(4 * HD);
  float* dsc    = alloc(4 * HD);
  unsigned* bar = (unsigned*)alloc(64);
  // bf16 weight cache (allocated as nElems/2 floats)
  auto allocbf = [&](size_t nelems) { return (unsigned short*)alloc(nelems / 2); };
  unsigned short* e0_whh_bf = allocbf((size_t)2 * G4 * HD);
  unsigned short* e1_whh_bf = allocbf((size_t)2 * G4 * HD);
  unsigned short* e1_wih_bf = allocbf((size_t)2 * G4 * 2 * HD);
  unsigned short* u_w_bf    = allocbf((size_t)HD * 2 * HD);
  unsigned short* d0_wih_bf = allocbf((size_t)2 * G4 * G4);
  unsigned short* d0_whh_bf = allocbf((size_t)2 * G4 * HD);
  unsigned short* d1_wih_bf = allocbf((size_t)2 * G4 * 2 * HD);
  unsigned short* d1_whh_bf = allocbf((size_t)2 * G4 * HD);
  unsigned short* h0out_bf  = allocbf((size_t)TS * 2 * HD);
  unsigned short* encout_bf = allocbf((size_t)TS * 2 * HD);
  (void)ws_size; (void)in_sizes; (void)n_in;

  k_init<<<1, 32, 0, stream>>>(bar);

  // Phase 0: one-time bf16 weight casts (~415 MB read, ~20 us at HBM rate)
  auto cast = [&](const float* s, unsigned short* d, size_t n) {
    k_tobf16<<<dim3(1024), NTH, 0, stream>>>(s, (unsigned*)d, (long)(n / 2));
  };
  cast(e0_whh, e0_whh_bf, (size_t)2 * G4 * HD);
  cast(e1_whh, e1_whh_bf, (size_t)2 * G4 * HD);
  cast(e1_wih, e1_wih_bf, (size_t)2 * G4 * 2 * HD);
  cast(u_w,    u_w_bf,    (size_t)HD * 2 * HD);
  cast(d0_wih, d0_wih_bf, (size_t)2 * G4 * G4);
  cast(d0_whh, d0_whh_bf, (size_t)2 * G4 * HD);
  cast(d1_wih, d1_wih_bf, (size_t)2 * G4 * 2 * HD);
  cast(d1_whh, d1_whh_bf, (size_t)2 * G4 * HD);

  // Phase A: layer-0 input projection (K=27, VALU)
  k_g0<<<dim3(2 * TS), NTH, 0, stream>>>(inputs, e0_wih, e0_bih, e0_bhh, g0);

  // Phase B: layer-0 bidirectional recurrence (persistent, 1 barrier/step)
  k_lstm_layer<<<NBLK, NTH, 0, stream>>>(g0, e0_whh_bf, h0out, h0T, gsc, bar, TS);
  cast(h0out, h0out_bf, (size_t)TS * 2 * HD);

  // Phase C: layer-1 input projection, both dirs: (256x2048)@(2048x4096)^T [bf16 WMMA]
  k_gemm_bf16<<<dim3(G4 / 16 / 8, TS / 16, 2), NTH, 0, stream>>>(
      h0out_bf, e1_wih_bf, e1_bih, e1_bhh, g1, TS, G4, 2 * HD,
      (long)G4 * (2 * HD), (long)G4, (long)TS * G4);

  // Phase D: layer-1 bidirectional recurrence
  k_lstm_layer<<<NBLK, NTH, 0, stream>>>(g1, e1_whh_bf, encout, h1T, gsc, bar, TS);
  cast(encout, encout_bf, (size_t)TS * 2 * HD);

  // Phase E: attention pre-projection u = enc @ u_w.T + u_b  [bf16 WMMA]
  k_gemm_bf16<<<dim3(HD / 16 / 8, TS / 16, 1), NTH, 0, stream>>>(
      encout_bf, u_w_bf, u_b, nullptr, uatt, TS, HD, 2 * HD, 0L, 0L, 0L);

  // Phase F: persistent attention decoder (bf16 d0/d1 weights)
  k_decoder<<<NBLK, NTH, 0, stream>>>(
      encout, uatt, h0T, h1T,
      w1_w, w1_b, w2_w, w2_b, w3_w, w3_b, w4_w, w4_b, w_w, w_b,
      att_w, att_b, o2c_w, o2c_b,
      d0_wih_bf, d0_whh_bf, d0_bih, d0_bhh,
      d1_wih_bf, d1_whh_bf, d1_bih, d1_bhh,
      ho_w, ho_b,
      dsh, dsc, wsum, wp, vatt,
      xt, out0g, outg, logits, decin, dgates, out, bar, steps);
}
